// InteractionBlock_82652350644685
// MI455X (gfx1250) — matrix operand order repaired and checked by reading, not compiled, not run on hardware
//
#include <hip/hip_runtime.h>
#include <hip/hip_bf16.h>
#include <math.h>

// ---------------------------------------------------------------------------
// CDNA5 (gfx1250) wave32 WMMA types
// ---------------------------------------------------------------------------
typedef __attribute__((ext_vector_type(16))) __bf16 bf16x16;
typedef __attribute__((ext_vector_type(8)))  float  f32x8;

union FragAB { bf16x16 v; uint4 u[2]; };

#define TILE_M 128
#define TILE_N 256
#define TILE_K 32
#define LDS_STRIDE 40   // 40 bf16 = 80B row stride: 16B-aligned rows, odd bank count

__device__ __forceinline__ float silu_f(float v) {
    return v / (1.0f + __expf(-v));
}

// Relaxed agent-scope fp32 atomic add -> global_atomic_add_f32 (no CAS loop)
__device__ __forceinline__ void atomic_add_f32(float* p, float v) {
    __hip_atomic_fetch_add(p, v, __ATOMIC_RELAXED, __HIP_MEMORY_SCOPE_AGENT);
}

// Pack 4 fp32 -> 4 bf16 and do one 8B LDS store
__device__ __forceinline__ void store_bf16x4(__bf16* dst, float4 v) {
    union { __bf16 h[4]; uint2 u; } pk;
    pk.h[0] = (__bf16)v.x; pk.h[1] = (__bf16)v.y;
    pk.h[2] = (__bf16)v.z; pk.h[3] = (__bf16)v.w;
    *reinterpret_cast<uint2*>(dst) = pk.u;
}

__device__ __forceinline__ void store_bf16x2(__bf16* dst, float2 v) {
    union { __bf16 h[2]; unsigned u; } pk;
    pk.h[0] = (__bf16)v.x; pk.h[1] = (__bf16)v.y;
    *reinterpret_cast<unsigned*>(dst) = pk.u;
}

// ---------------------------------------------------------------------------
// Generic tiled GEMM:  C = epilogue( A@B [+ A2@B2] + bias ),  bf16 WMMA, f32 acc
// Block tile 128x256, 8 waves of 64x64 (4x4 WMMA fragments each).
// Requirements (true for every call site): K % 32 == 0, N == 256.
// A:[M,K] B:[K,N] C:[M,N], row-major fp32.
// epilogue: v = acc + bias; if(act) v = silu(v); if(resid) v += resid[row,col]
// ---------------------------------------------------------------------------
__global__ __launch_bounds__(256)
void gemm_bf16_wmma(const float* __restrict__ A,  const float* __restrict__ B,
                    const float* __restrict__ A2, const float* __restrict__ B2,
                    const float* __restrict__ bias, const float* __restrict__ resid,
                    float* __restrict__ C,
                    int M, int N, int K, int K2, int act)
{
    __shared__ alignas(16) __bf16 As[TILE_M][LDS_STRIDE];
    __shared__ alignas(16) __bf16 Bs[TILE_N][LDS_STRIDE];

    const int tid   = threadIdx.x;
    const int lane  = tid & 31;
    const int wid   = tid >> 5;      // 8 waves
    const int waveM = wid >> 2;      // 0..1 -> 64-row band
    const int waveN = wid & 3;       // 0..3 -> 64-col band
    const int mrow  = lane & 15;
    const int hi    = lane >> 4;     // lane group (0: lanes 0-15, 1: lanes 16-31)

    const int rowBase = blockIdx.y * TILE_M;
    const int colBase = blockIdx.x * TILE_N;

    f32x8 acc[4][4];
    #pragma unroll
    for (int i = 0; i < 4; ++i)
        #pragma unroll
        for (int j = 0; j < 4; ++j)
            #pragma unroll
            for (int r = 0; r < 8; ++r) acc[i][j][r] = 0.0f;

    const int npass = (A2 != nullptr) ? 2 : 1;
    for (int pass = 0; pass < npass; ++pass) {
        const float* Ap = pass ? A2 : A;
        const float* Bp = pass ? B2 : B;
        const int    Kp = pass ? K2 : K;

        for (int k0 = 0; k0 < Kp; k0 += TILE_K) {
            // ---- branch-free vector loads (all in flight, no exec branching)
            float4 av[4];
            int ar[4], ac4[4];
            #pragma unroll
            for (int i = 0; i < 4; ++i) {
                int idx4 = tid + i * 256;          // 1024 float4 = 128x32
                ar[i]  = idx4 >> 3;                // row 0..127
                ac4[i] = (idx4 & 7) << 2;          // col 0,4,..,28
                int gr  = rowBase + ar[i];
                int grc = min(gr, M - 1);
                float4 v = *reinterpret_cast<const float4*>(
                    Ap + (size_t)(unsigned)(grc * Kp + k0 + ac4[i]));
                if (gr >= M) v = make_float4(0.f, 0.f, 0.f, 0.f);
                av[i] = v;
            }
            float4 bv[8];
            int bk[8], bc4[8];
            #pragma unroll
            for (int i = 0; i < 8; ++i) {
                int idx4 = tid + i * 256;          // 2048 float4 = 32x256
                bk[i]  = idx4 >> 6;                // k row 0..31
                bc4[i] = (idx4 & 63) << 2;         // col 0..252
                bv[i] = *reinterpret_cast<const float4*>(
                    Bp + (size_t)(unsigned)((k0 + bk[i]) * N + colBase + bc4[i]));
            }
            // prefetch next K tile while we convert/compute
            int nk = k0 + TILE_K;
            if (nk < Kp) {
                __builtin_prefetch(Ap + (size_t)(unsigned)(min(rowBase + (tid >> 3), M - 1) * Kp + nk + ((tid & 7) << 2)), 0, 1);
                __builtin_prefetch(Bp + (size_t)(unsigned)((nk + (tid >> 6)) * N + colBase + ((tid & 63) << 2)), 0, 1);
            }
            // ---- convert + LDS store (A direct, B transposed)
            #pragma unroll
            for (int i = 0; i < 4; ++i)
                store_bf16x4(&As[ar[i]][ac4[i]], av[i]);
            #pragma unroll
            for (int i = 0; i < 8; ++i) {
                Bs[bc4[i] + 0][bk[i]] = (__bf16)bv[i].x;
                Bs[bc4[i] + 1][bk[i]] = (__bf16)bv[i].y;
                Bs[bc4[i] + 2][bk[i]] = (__bf16)bv[i].z;
                Bs[bc4[i] + 3][bk[i]] = (__bf16)bv[i].w;
            }
            __syncthreads();

            // A fragment per ISA 16-bit 16x32 layout:
            //   lanes 0-15: K {0..7,16..23}, lanes 16-31: K {8..15,24..31}
            FragAB af[4];
            #pragma unroll
            for (int fm = 0; fm < 4; ++fm) {
                const __bf16* ap = &As[waveM * 64 + fm * 16 + mrow][hi * 8];
                af[fm].u[0] = *reinterpret_cast<const uint4*>(ap);
                af[fm].u[1] = *reinterpret_cast<const uint4*>(ap + 16);
            }
            #pragma unroll
            for (int fn = 0; fn < 4; ++fn) {
                // B fragment: lanes 0-15 hold K {0..7,16..23}, lanes 16-31 hold K {8..15,24..31}
                FragAB bfv;
                const __bf16* bp = &Bs[waveN * 64 + fn * 16 + mrow][hi * 8];
                bfv.u[0] = *reinterpret_cast<const uint4*>(bp);
                bfv.u[1] = *reinterpret_cast<const uint4*>(bp + 16);
                #pragma unroll
                for (int fm = 0; fm < 4; ++fm) {
                    acc[fm][fn] = __builtin_amdgcn_wmma_f32_16x16x32_bf16(
                        false, af[fm].v, false, bfv.v, (short)0, acc[fm][fn],
                        false, false);
                }
            }
            __syncthreads();
        }
    }

    // Epilogue (C/D layout: VGPR r -> M = r + 8*hi, lane%16 -> N)
    #pragma unroll
    for (int fm = 0; fm < 4; ++fm) {
        #pragma unroll
        for (int fn = 0; fn < 4; ++fn) {
            int col = colBase + waveN * 64 + fn * 16 + mrow;
            float bv2 = (bias != nullptr) ? bias[col] : 0.0f;
            #pragma unroll
            for (int r = 0; r < 8; ++r) {
                int row = rowBase + waveM * 64 + fm * 16 + hi * 8 + r;
                if (row < M) {
                    float v = acc[fm][fn][r] + bv2;
                    if (act)   v = silu_f(v);
                    if (resid) v += resid[(size_t)(unsigned)(row * N + col)];
                    C[(size_t)(unsigned)(row * N + col)] = v;
                }
            }
        }
    }
}

// ---------------------------------------------------------------------------
// Fused edge conv:  for each edge tile, C = featTile @ Wf (WMMA), then
// msg = C * x1[src[e]] and atomicAdd into agg[dst[e]].  f-matrices never hit
// HBM, and with TILE_N == H each feature row is streamed exactly once.
// Requirements: F even (rows of feat are 8B-aligned), Hh == 256 == TILE_N.
// ---------------------------------------------------------------------------
__global__ __launch_bounds__(256)
void edge_conv_wmma(const float* __restrict__ feat, const float* __restrict__ Wf,
                    const float* __restrict__ x1,
                    const int* __restrict__ srcIdx, const int* __restrict__ dstIdx,
                    float* __restrict__ agg, int E, int F, int Hh)
{
    __shared__ alignas(16) __bf16 As[TILE_M][LDS_STRIDE];
    __shared__ alignas(16) __bf16 Bs[TILE_N][LDS_STRIDE];
    __shared__ int sSrc[TILE_M];
    __shared__ int sDst[TILE_M];

    const int tid   = threadIdx.x;
    const int lane  = tid & 31;
    const int wid   = tid >> 5;
    const int waveM = wid >> 2;
    const int waveN = wid & 3;
    const int mrow  = lane & 15;
    const int hi    = lane >> 4;

    const int eBase   = blockIdx.y * TILE_M;
    const int colBase = blockIdx.x * TILE_N;   // always 0 (one block spans H)

    if (tid < TILE_M) {
        int e = eBase + tid;
        int ec = min(e, E - 1);
        sSrc[tid] = srcIdx[ec];
        sDst[tid] = dstIdx[ec];
    }

    f32x8 acc[4][4];
    #pragma unroll
    for (int i = 0; i < 4; ++i)
        #pragma unroll
        for (int j = 0; j < 4; ++j)
            #pragma unroll
            for (int r = 0; r < 8; ++r) acc[i][j][r] = 0.0f;

    for (int k0 = 0; k0 < F; k0 += TILE_K) {
        // feature tile: float2 loads (F even -> 8B aligned), clamp + select-zero
        float2 av[8];
        int ar[8], ac2[8];
        #pragma unroll
        for (int i = 0; i < 8; ++i) {
            int idx2 = tid + i * 256;              // 2048 float2 = 128x32
            ar[i]  = idx2 >> 4;                    // row 0..127
            ac2[i] = (idx2 & 15) << 1;             // col 0,2,..,30
            int ge  = eBase + ar[i];
            int gec = min(ge, E - 1);
            int gc  = k0 + ac2[i];
            int gcc = min(gc, F - 2);
            float2 v = *reinterpret_cast<const float2*>(
                feat + (size_t)(unsigned)(gec * F + gcc));
            float vx = (ge < E && gc     < F) ? v.x : 0.0f;
            float vy = (ge < E && gc + 1 < F) ? v.y : 0.0f;
            av[i] = make_float2(vx, vy);
        }
        // Wf tile: float4 loads along N (32x256), clamp k to F-1 + select-zero
        float4 bv[8];
        int bk[8], bc4[8];
        #pragma unroll
        for (int i = 0; i < 8; ++i) {
            int idx4 = tid + i * 256;
            bk[i]  = idx4 >> 6;
            bc4[i] = (idx4 & 63) << 2;
            int gk  = k0 + bk[i];
            int gkc = min(gk, F - 1);
            float4 v = *reinterpret_cast<const float4*>(
                Wf + (size_t)(unsigned)(gkc * Hh + colBase + bc4[i]));
            if (gk >= F) v = make_float4(0.f, 0.f, 0.f, 0.f);
            bv[i] = v;
        }
        int nk = k0 + TILE_K;
        if (nk < F) {
            __builtin_prefetch(feat + (size_t)(unsigned)(min(eBase + (tid >> 4), E - 1) * F + min(nk + ((tid & 15) << 1), F - 2)), 0, 1);
        }
        #pragma unroll
        for (int i = 0; i < 8; ++i)
            store_bf16x2(&As[ar[i]][ac2[i]], av[i]);
        #pragma unroll
        for (int i = 0; i < 8; ++i) {
            Bs[bc4[i] + 0][bk[i]] = (__bf16)bv[i].x;
            Bs[bc4[i] + 1][bk[i]] = (__bf16)bv[i].y;
            Bs[bc4[i] + 2][bk[i]] = (__bf16)bv[i].z;
            Bs[bc4[i] + 3][bk[i]] = (__bf16)bv[i].w;
        }
        __syncthreads();

        FragAB af[4];
        #pragma unroll
        for (int fm = 0; fm < 4; ++fm) {
            const __bf16* ap = &As[waveM * 64 + fm * 16 + mrow][hi * 8];
            af[fm].u[0] = *reinterpret_cast<const uint4*>(ap);
            af[fm].u[1] = *reinterpret_cast<const uint4*>(ap + 16);
        }
        #pragma unroll
        for (int fn = 0; fn < 4; ++fn) {
            FragAB bfv;
            const __bf16* bp = &Bs[waveN * 64 + fn * 16 + mrow][hi * 8];
            bfv.u[0] = *reinterpret_cast<const uint4*>(bp);
            bfv.u[1] = *reinterpret_cast<const uint4*>(bp + 16);
            #pragma unroll
            for (int fm = 0; fm < 4; ++fm) {
                acc[fm][fn] = __builtin_amdgcn_wmma_f32_16x16x32_bf16(
                    false, af[fm].v, false, bfv.v, (short)0, acc[fm][fn],
                    false, false);
            }
        }
        __syncthreads();
    }

    // Epilogue: gather x1[src], Hadamard, scatter-add to agg[dst]
    #pragma unroll
    for (int fm = 0; fm < 4; ++fm) {
        #pragma unroll
        for (int fn = 0; fn < 4; ++fn) {
            int col = colBase + waveN * 64 + fn * 16 + mrow;
            #pragma unroll
            for (int r = 0; r < 8; ++r) {
                int re = waveM * 64 + fm * 16 + hi * 8 + r;   // edge row in tile
                int e  = eBase + re;
                if (e < E) {
                    int s = sSrc[re];
                    int d = sDst[re];
                    float msg = acc[fm][fn][r] * x1[(size_t)(unsigned)(s * Hh + col)];
                    atomic_add_f32(&agg[(size_t)(unsigned)(d * Hh + col)], msg);
                }
            }
        }
    }
}

// ---------------------------------------------------------------------------
// Small helper kernels
// ---------------------------------------------------------------------------
__global__ void zero_kernel(float* __restrict__ p, size_t n) {
    size_t i = (size_t)blockIdx.x * 256 + threadIdx.x;
    if (i < n) p[i] = 0.0f;
}

__global__ void gn_accum(const float* __restrict__ h, const int* __restrict__ batch,
                         float* __restrict__ gsum, float* __restrict__ gcnt,
                         int Nn, int Hh) {
    size_t i = (size_t)blockIdx.x * 256 + threadIdx.x;
    if (i >= (size_t)Nn * Hh) return;
    int n = (int)(i / Hh), c = (int)(i % Hh);
    int g = batch[n];
    atomic_add_f32(&gsum[(size_t)(unsigned)(g * Hh + c)], h[i]);
    if (c == 0) atomic_add_f32(&gcnt[g], 1.0f);
}

__global__ void gn_center(const float* __restrict__ h, const int* __restrict__ batch,
                          const float* __restrict__ gsum, const float* __restrict__ gcnt,
                          const float* __restrict__ ms,
                          float* __restrict__ out, float* __restrict__ gvar,
                          int Nn, int Hh) {
    size_t i = (size_t)blockIdx.x * 256 + threadIdx.x;
    if (i >= (size_t)Nn * Hh) return;
    int n = (int)(i / Hh), c = (int)(i % Hh);
    int g = batch[n];
    float cnt = fmaxf(gcnt[g], 1.0f);
    float o = h[i] - (gsum[(size_t)(unsigned)(g * Hh + c)] / cnt) * ms[c];
    out[i] = o;
    atomic_add_f32(&gvar[(size_t)(unsigned)(g * Hh + c)], o * o);
}

__global__ void gn_final(float* __restrict__ out, const int* __restrict__ batch,
                         const float* __restrict__ gvar, const float* __restrict__ gcnt,
                         const float* __restrict__ w, const float* __restrict__ b,
                         int Nn, int Hh) {
    size_t i = (size_t)blockIdx.x * 256 + threadIdx.x;
    if (i >= (size_t)Nn * Hh) return;
    int n = (int)(i / Hh), c = (int)(i % Hh);
    int g = batch[n];
    float cnt = fmaxf(gcnt[g], 1.0f);
    float v = gvar[(size_t)(unsigned)(g * Hh + c)] / cnt;
    out[i] = w[c] * out[i] * rsqrtf(v + 1e-5f) + b[c];
}

// ---------------------------------------------------------------------------
// Host-side orchestration
// ---------------------------------------------------------------------------
extern "C" void kernel_launch(void* const* d_in, const int* in_sizes, int n_in,
                              void* d_out, int out_size, void* d_ws, size_t ws_size,
                              hipStream_t stream)
{
    const int H = 256, G = 512, Lnum = 3, F1 = 294, F2 = 42;
    const int Nn = in_sizes[0] / H;
    const int E  = in_sizes[1] / F1;

    const float* x        = (const float*)d_in[0];
    const float* feature1 = (const float*)d_in[1];
    const float* feature2 = (const float*)d_in[2];
    const int*   edge_idx = (const int*)d_in[3];
    const int*   batch    = (const int*)d_in[4];
    const float* W_lin = (const float*)d_in[5];
    const float* b_lin = (const float*)d_in[6];
    const float* Wf1_1 = (const float*)d_in[7];
    const float* Wf1_2 = (const float*)d_in[8];
    const float* Wf2_1 = (const float*)d_in[9];
    const float* Wf2_2 = (const float*)d_in[10];
    const float* Wl1 = (const float*)d_in[11];
    const float* bl1 = (const float*)d_in[12];
    const float* Wr1 = (const float*)d_in[13];
    const float* Wl2 = (const float*)d_in[14];
    const float* bl2 = (const float*)d_in[15];
    const float* Wr2 = (const float*)d_in[16];
    const float* W1  = (const float*)d_in[17];
    const float* b1  = (const float*)d_in[18];
    const float* W2  = (const float*)d_in[19];
    const float* b2  = (const float*)d_in[20];
    const float* Wcat  = (const float*)d_in[21];
    const float* bcat  = (const float*)d_in[22];
    const float* Wlins = (const float*)d_in[23];
    const float* blins = (const float*)d_in[24];
    const float* g_w  = (const float*)d_in[25];
    const float* g_b  = (const float*)d_in[26];
    const float* g_ms = (const float*)d_in[27];
    const float* Wfin = (const float*)d_in[28];
    const float* bfin = (const float*)d_in[29];

    const int* srcIdx = edge_idx;
    const int* dstIdx = edge_idx + E;

    const size_t NH = (size_t)Nn * H;
    float* ws    = (float*)d_ws;
    float* Wf1c  = ws;                         // F1*H  (combined Wf1_1@Wf1_2)
    float* Wf2c  = Wf1c + (size_t)F1 * H;      // F2*H
    float* x1    = Wf2c + (size_t)F2 * H;      // N*H   silu(x@W_lin+b)
    float* bufA  = x1   + NH;                  // N*H   agg1 -> h -> ping-pong
    float* bufB  = bufA + NH;                  // N*H   agg2 -> ping-pong
    float* bufT  = bufB + NH;                  // N*H   temp / normed h
    float* bufH1 = bufT + NH;                  // N*H
    float* bufH2 = bufH1 + NH;                 // N*H
    float* gsum  = bufH2 + NH;                 // G*H
    float* gvar  = gsum + (size_t)G * H;       // G*H
    float* gcnt  = gvar + (size_t)G * H;       // G

    auto cdiv = [](int a, int b) { return (a + b - 1) / b; };

    // Zero atomic accumulators (bufA,bufB contiguous; gsum,gvar,gcnt contiguous)
    {
        size_t nz = 2 * NH;
        zero_kernel<<<dim3((unsigned)((nz + 255) / 256)), 256, 0, stream>>>(bufA, nz);
        size_t nz2 = 2 * (size_t)G * H + G;
        zero_kernel<<<dim3((unsigned)((nz2 + 255) / 256)), 256, 0, stream>>>(gsum, nz2);
    }

    #define GEMM(Ap, Bp, A2p, B2p, biasp, residp, Cp, Mv, Kv, K2v, actv)            \
        gemm_bf16_wmma<<<dim3(cdiv(H, TILE_N), cdiv((Mv), TILE_M)), 256, 0, stream>>>( \
            (Ap), (Bp), (A2p), (B2p), (biasp), (residp), (Cp), (Mv), H, (Kv), (K2v), (actv))

    // Combined per-edge weight matrices: Wf = Wf_1 @ Wf_2 (re-association)
    GEMM(Wf1_1, Wf1_2, nullptr, nullptr, nullptr, nullptr, Wf1c, F1, H, 0, 0);
    GEMM(Wf2_1, Wf2_2, nullptr, nullptr, nullptr, nullptr, Wf2c, F2, H, 0, 0);

    // x1 = silu(x @ W_lin + b_lin)
    GEMM(x, W_lin, nullptr, nullptr, b_lin, nullptr, x1, Nn, H, 0, 1);

    // Fused edge conv (GEMM + gather + Hadamard + scatter-add), both branches
    edge_conv_wmma<<<dim3(cdiv(H, TILE_N), cdiv(E, TILE_M)), 256, 0, stream>>>(
        feature1, Wf1c, x1, srcIdx, dstIdx, bufA, E, F1, H);
    edge_conv_wmma<<<dim3(cdiv(H, TILE_N), cdiv(E, TILE_M)), 256, 0, stream>>>(
        feature2, Wf2c, x1, srcIdx, dstIdx, bufB, E, F2, H);

    // Branch 1: t = agg1@Wl1 + x1@Wr1 + bl1 ; h1 = silu(t@W1 + b1)
    GEMM(bufA, Wl1, x1, Wr1, bl1, nullptr, bufT, Nn, H, H, 0);
    GEMM(bufT, W1, nullptr, nullptr, b1, nullptr, bufH1, Nn, H, 0, 1);
    // Branch 2
    GEMM(bufB, Wl2, x1, Wr2, bl2, nullptr, bufT, Nn, H, H, 0);
    GEMM(bufT, W2, nullptr, nullptr, b2, nullptr, bufH2, Nn, H, 0, 1);

    // h = h1@Wcat_top + h2@Wcat_bot + bcat + x1
    GEMM(bufH1, Wcat, bufH2, Wcat + (size_t)H * H, bcat, x1, bufA, Nn, H, H, 0);

    // Residual MLP x3: h = silu(h@W + b) + h   (ping-pong bufA <-> bufB)
    float* hin = bufA;
    float* hout = bufB;
    for (int i = 0; i < Lnum; ++i) {
        GEMM(hin, Wlins + (size_t)i * H * H, nullptr, nullptr,
             blins + (size_t)i * H, hin, hout, Nn, H, 0, 1);
        float* t = hin; hin = hout; hout = t;
    }

    // GraphNorm on hin -> bufT
    {
        size_t tot = NH;
        unsigned gb = (unsigned)((tot + 255) / 256);
        gn_accum<<<dim3(gb), 256, 0, stream>>>(hin, batch, gsum, gcnt, Nn, H);
        gn_center<<<dim3(gb), 256, 0, stream>>>(hin, batch, gsum, gcnt, g_ms, bufT, gvar, Nn, H);
        gn_final<<<dim3(gb), 256, 0, stream>>>(bufT, batch, gvar, gcnt, g_w, g_b, Nn, H);
    }

    // out = h_norm @ Wfin + bfin
    GEMM(bufT, Wfin, nullptr, nullptr, bfin, nullptr, (float*)d_out, Nn, H, 0, 0);

    #undef GEMM
}
